// NNUE_16441134809368
// MI455X (gfx1250) — matrix operand order repaired and checked by reading, not compiled, not run on hardware
//
#include <hip/hip_runtime.h>
#include <hip/hip_bf16.h>

typedef __attribute__((ext_vector_type(16))) _Float16 v16h;
typedef __attribute__((ext_vector_type(8)))  _Float16 v8h;
typedef __attribute__((ext_vector_type(8)))  float    v8f;
typedef __attribute__((ext_vector_type(4)))  float    v4f;

#define NB     4096
#define FTSZ   2048
#define ROW    2056   // FT + BK
#define MAXF   32
#define TWOFT  4096   // 2*FT == K of the L1 GEMM
#define NOUT   256    // HID*BK == N of the L1 GEMM
#define KSTEP  32
#define NKSTEPS (TWOFT / KSTEP)   // 128 (even)

#ifndef GEMM_ASYNC
#define GEMM_ASYNC 1
#endif

// ---------------------------------------------------------------------------
// helpers
// ---------------------------------------------------------------------------
__device__ __forceinline__ v8f wmma_f16(v16h a, v16h b, v8f c) {
    return __builtin_amdgcn_wmma_f32_16x16x32_f16(
        false, a, false, b, (short)0, c, false, false);
}

__device__ __forceinline__ void stage16(_Float16* lptr, const _Float16* gptr) {
#if GEMM_ASYNC
    // 16-byte async copy global -> LDS, tracked by ASYNCcnt.
    // Low 32 bits of a generic LDS pointer are the LDS byte offset (ISA 10.2).
    asm volatile("global_load_async_to_lds_b128 %0, %1, off"
                 :: "v"((unsigned int)(unsigned long long)lptr), "v"(gptr)
                 : "memory");
#else
    *(v8h*)lptr = *(const v8h*)gptr;
#endif
}

__device__ __forceinline__ void stage_wait() {
#if GEMM_ASYNC
    asm volatile("s_wait_asynccnt 0" ::: "memory");
#endif
}

union Frag { v16h v; v8h h[2]; };

// ---------------------------------------------------------------------------
// Kernel 0: convert l1_w (256x4096 fp32, row-major N x K) to fp16, 4-wide
// ---------------------------------------------------------------------------
__global__ void cvt_f32_to_f16(const float* __restrict__ src,
                               _Float16* __restrict__ dst, int n4) {
    int i = blockIdx.x * blockDim.x + threadIdx.x;
    if (i < n4) {
        v4f s = *(const v4f*)(src + i * 4);
        _Float16* d = dst + i * 4;
        d[0] = (_Float16)s[0]; d[1] = (_Float16)s[1];
        d[2] = (_Float16)s[2]; d[3] = (_Float16)s[3];
    }
}

// ---------------------------------------------------------------------------
// Kernel 1: feature transformer. One block (256 threads) per sample.
// ft_w = 185 MB < 192 MB L2, so the ~2.1 GB of gathers hit L2 after the first
// streaming pass. Each thread owns 8 CONTIGUOUS columns -> b128 load pairs,
// v8h packed fp16 stores, and global_prefetch of the next feature row.
// ---------------------------------------------------------------------------
__global__ void __launch_bounds__(256)
ft_kernel(const int* __restrict__ wf, const int* __restrict__ bfeat,
          const float* __restrict__ stm, const int* __restrict__ pc,
          const float* __restrict__ ft_w, const float* __restrict__ ft_b,
          _Float16* __restrict__ l1in, float* __restrict__ psqt) {
    __shared__ int   sW[MAXF];
    __shared__ int   sB[MAXF];
    __shared__ float sred[2];

    const int b   = blockIdx.x;
    const int tid = threadIdx.x;

    if (tid < MAXF)          sW[tid]        = wf[b * MAXF + tid];
    else if (tid < 2 * MAXF) sB[tid - MAXF] = bfeat[b * MAXF + (tid - MAXF)];
    __syncthreads();

    const int bucket = (pc[b] - 1) >> 2;

    // PSQT column (FT + bucket): waves 0 (white) and 1 (black), lane = feature
    if (tid < 64) {
        const int  lane = tid & 31;
        const int* idxs = (tid < 32) ? sW : sB;
        const int  fi   = idxs[lane];
        float v = (fi >= 0) ? ft_w[(size_t)fi * ROW + FTSZ + bucket] : 0.0f;
        #pragma unroll
        for (int off = 16; off > 0; off >>= 1) v += __shfl_xor(v, off, 32);
        if (lane == 0) sred[tid >> 5] = v;
    }

    const int colbase = tid * 8;           // 8 contiguous columns per thread
    v4f wacc0 = {0.f, 0.f, 0.f, 0.f}, wacc1 = {0.f, 0.f, 0.f, 0.f};
    v4f bacc0 = {0.f, 0.f, 0.f, 0.f}, bacc1 = {0.f, 0.f, 0.f, 0.f};

    for (int i = 0; i < MAXF; i++) {
        const int fi = sW[i];
        if (fi < 0) break;                 // valid indices are a prefix
        const int nf = (i + 1 < MAXF) ? sW[i + 1] : -1;
        if (nf >= 0) __builtin_prefetch(ft_w + (size_t)nf * ROW + colbase, 0, 0);
        const float* r = ft_w + (size_t)fi * ROW + colbase;
        wacc0 += *(const v4f*)r;
        wacc1 += *(const v4f*)(r + 4);
    }
    for (int i = 0; i < MAXF; i++) {
        const int fi = sB[i];
        if (fi < 0) break;
        const int nf = (i + 1 < MAXF) ? sB[i + 1] : -1;
        if (nf >= 0) __builtin_prefetch(ft_w + (size_t)nf * ROW + colbase, 0, 0);
        const float* r = ft_w + (size_t)fi * ROW + colbase;
        bacc0 += *(const v4f*)r;
        bacc1 += *(const v4f*)(r + 4);
    }

    const float s = stm[b];
    const bool  white_first = (s > 0.5f);
    const v4f fb0 = *(const v4f*)(ft_b + colbase);
    const v4f fb1 = *(const v4f*)(ft_b + colbase + 4);

    v8h fr, bk;
    #pragma unroll
    for (int j = 0; j < 8; j++) {
        const float wa = (j < 4) ? wacc0[j & 3] + fb0[j & 3] : wacc1[j & 3] + fb1[j & 3];
        const float ba = (j < 4) ? bacc0[j & 3] + fb0[j & 3] : bacc1[j & 3] + fb1[j & 3];
        const float wv = fminf(fmaxf(wa, 0.f), 1.f);
        const float bv = fminf(fmaxf(ba, 0.f), 1.f);
        fr[j] = (_Float16)(white_first ? wv : bv);
        bk[j] = (_Float16)(white_first ? bv : wv);
    }
    _Float16* out = l1in + (size_t)b * TWOFT;
    *(v8h*)(out + colbase)        = fr;    // one 16B store
    *(v8h*)(out + FTSZ + colbase) = bk;    // one 16B store

    __syncthreads();
    if (tid == 0) psqt[b] = (sred[0] - sred[1]) * (s - 0.5f);
}

// ---------------------------------------------------------------------------
// Kernel 2: L1 GEMM, C[4096x256] = A[4096x4096] * B[4096x256] (fp16 in, fp32
// acc). One block per 64-row group; 8 waves cover the full N=256 (2 n-tiles
// each). A tiles are double-buffered in LDS via global_load_async_to_lds_b128
// so A is read from global exactly once. The k-loop is unrolled x2 with
// even/odd register sets, so the B double-buffer rotation needs NO register
// copies (the v_mov_b64 storm + hazard v_nops from round 2 disappear). All 4
// A fragments are loaded before the 8 WMMAs of a phase so ds_load writes and
// fragment consumption are separated by a full WMMA burst.
// ---------------------------------------------------------------------------
__global__ void __launch_bounds__(256)
gemm_l1_wmma(const _Float16* __restrict__ A, const _Float16* __restrict__ Bm,
             float* __restrict__ C) {
    __shared__ _Float16 lA[2][64 * KSTEP];      // 2 x 4 KB

    const int tid   = threadIdx.x;
    const int wv    = tid >> 5;
    const int lane  = tid & 31;
    const int lr    = lane & 15;
    const int lh    = lane >> 4;
    const int mbase = blockIdx.x * 64;
    const int nbase = wv * 32;

    // staging: each of 256 threads copies 16 bytes of the 64x32 tile
    const int srow = tid >> 2;                  // 0..63
    const int scol = (tid & 3) * 8;             // 0,8,16,24
    const _Float16* gstage  = A + (size_t)(mbase + srow) * TWOFT + scol;
    _Float16* lstage0 = &lA[0][srow * KSTEP + scol];
    _Float16* lstage1 = &lA[1][srow * KSTEP + scol];

    const _Float16* bptr0 = Bm + (size_t)(nbase + lr) * TWOFT + lh * 16;
    const _Float16* bptr1 = bptr0 + (size_t)16 * TWOFT;

    v8f acc[4][2] = {};

    // prologue: stage k=0 tile, preload even-phase B fragments
    stage16(lstage0, gstage);
    Frag bE0, bE1, bO0, bO1;
    bE0.h[0] = *(const v8h*)(bptr0);  bE0.h[1] = *(const v8h*)(bptr0 + 8);
    bE1.h[0] = *(const v8h*)(bptr1);  bE1.h[1] = *(const v8h*)(bptr1 + 8);
    stage_wait();
    __syncthreads();

    for (int ks = 0; ks < NKSTEPS; ks += 2) {
        // ---------- even phase: consume lA[0]/bE, prepare lA[1]/bO ----------
        {
            const int kn = (ks + 1) * KSTEP;
            stage16(lstage1, gstage + kn);
            bO0.h[0] = *(const v8h*)(bptr0 + kn);
            bO0.h[1] = *(const v8h*)(bptr0 + kn + 8);
            bO1.h[0] = *(const v8h*)(bptr1 + kn);
            bO1.h[1] = *(const v8h*)(bptr1 + kn + 8);

            Frag af[4];
            #pragma unroll
            for (int t = 0; t < 4; ++t) {
                const _Float16* ap = &lA[0][(t * 16 + lr) * KSTEP + lh * 16];
                af[t].h[0] = *(const v8h*)ap;
                af[t].h[1] = *(const v8h*)(ap + 8);
            }
            #pragma unroll
            for (int t = 0; t < 4; ++t) {
                acc[t][0] = wmma_f16(af[t].v, bE0.v, acc[t][0]);
                acc[t][1] = wmma_f16(af[t].v, bE1.v, acc[t][1]);
            }
            stage_wait();
            __syncthreads();
        }
        // ---------- odd phase: consume lA[1]/bO, prepare lA[0]/bE ----------
        {
            const bool more = (ks + 2 < NKSTEPS);
            const int kn = (ks + 2) * KSTEP;
            if (more) {
                stage16(lstage0, gstage + kn);
                bE0.h[0] = *(const v8h*)(bptr0 + kn);
                bE0.h[1] = *(const v8h*)(bptr0 + kn + 8);
                bE1.h[0] = *(const v8h*)(bptr1 + kn);
                bE1.h[1] = *(const v8h*)(bptr1 + kn + 8);
            }

            Frag af[4];
            #pragma unroll
            for (int t = 0; t < 4; ++t) {
                const _Float16* ap = &lA[1][(t * 16 + lr) * KSTEP + lh * 16];
                af[t].h[0] = *(const v8h*)ap;
                af[t].h[1] = *(const v8h*)(ap + 8);
            }
            #pragma unroll
            for (int t = 0; t < 4; ++t) {
                acc[t][0] = wmma_f16(af[t].v, bO0.v, acc[t][0]);
                acc[t][1] = wmma_f16(af[t].v, bO1.v, acc[t][1]);
            }
            stage_wait();
            __syncthreads();
        }
    }

    // C 16x16 f32 frag: VGPR r -> row r (lanes 0-15) / row 8+r (lanes 16-31)
    #pragma unroll
    for (int t = 0; t < 4; ++t)
        #pragma unroll
        for (int nn = 0; nn < 2; ++nn)
            #pragma unroll
            for (int r = 0; r < 8; ++r) {
                const int row = mbase + t * 16 + lh * 8 + r;
                const int col = nbase + nn * 16 + lr;
                C[(size_t)row * NOUT + col] = acc[t][nn][r];
            }
}

// ---------------------------------------------------------------------------
// Kernel 3: head. One wave per sample: bucket select + l1 bias, l2 (32x32
// matvec via shuffle-broadcast), clip, out dot via xor-reduce, + bias + psqt.
// ---------------------------------------------------------------------------
__global__ void __launch_bounds__(256)
head_kernel(const float* __restrict__ l1out, const float* __restrict__ l1_b,
            const float* __restrict__ l2_w,  const float* __restrict__ l2_b,
            const float* __restrict__ out_w, const float* __restrict__ out_b,
            const int* __restrict__ pc, const float* __restrict__ psqt,
            float* __restrict__ outp) {
    const int b = blockIdx.x * (blockDim.x >> 5) + (threadIdx.x >> 5);
    if (b >= NB) return;
    const int lane   = threadIdx.x & 31;
    const int bucket = (pc[b] - 1) >> 2;
    const int base   = bucket * 32;

    const float l1sel = l1out[(size_t)b * NOUT + base + lane] + l1_b[base + lane];

    float acc = l2_b[base + lane];
    const float* wrow = l2_w + (size_t)(base + lane) * 32;
    #pragma unroll 8
    for (int k = 0; k < 32; k++) acc += wrow[k] * __shfl(l1sel, k, 32);
    acc = fminf(fmaxf(acc, 0.f), 1.f);

    float term = acc * out_w[base + lane];
    #pragma unroll
    for (int off = 16; off > 0; off >>= 1) term += __shfl_xor(term, off, 32);

    if (lane == 0) outp[b] = term + out_b[bucket] + psqt[b];
}

// ---------------------------------------------------------------------------
extern "C" void kernel_launch(void* const* d_in, const int* in_sizes, int n_in,
                              void* d_out, int out_size, void* d_ws, size_t ws_size,
                              hipStream_t stream) {
    const int*   wf    = (const int*)  d_in[0];
    const int*   bfeat = (const int*)  d_in[1];
    const float* stm   = (const float*)d_in[2];
    const int*   pc    = (const int*)  d_in[3];
    const float* ft_w  = (const float*)d_in[4];
    const float* ft_b  = (const float*)d_in[5];
    const float* l1_w  = (const float*)d_in[6];
    const float* l1_b  = (const float*)d_in[7];
    const float* l2_w  = (const float*)d_in[8];
    const float* l2_b  = (const float*)d_in[9];
    const float* out_w = (const float*)d_in[10];
    const float* out_b = (const float*)d_in[11];
    float* out = (float*)d_out;

    char* ws = (char*)d_ws;
    _Float16* l1in_h = (_Float16*)ws;                                  // 32 MB
    _Float16* l1w_h  = (_Float16*)(ws + (size_t)32 * 1024 * 1024);     //  2 MB
    float*    l1out  = (float*)   (ws + (size_t)34 * 1024 * 1024);     //  4 MB
    float*    psqt   = (float*)   (ws + (size_t)38 * 1024 * 1024);     // 16 KB

    cvt_f32_to_f16<<<(NOUT * TWOFT / 4 + 255) / 256, 256, 0, stream>>>(
        l1_w, l1w_h, NOUT * TWOFT / 4);
    ft_kernel<<<NB, 256, 0, stream>>>(wf, bfeat, stm, pc, ft_w, ft_b,
                                      l1in_h, psqt);
    gemm_l1_wmma<<<TWOFT / 64, 256, 0, stream>>>(l1in_h, l1w_h, l1out);
    head_kernel<<<NB / 8, 256, 0, stream>>>(l1out, l1_b, l2_w, l2_b,
                                            out_w, out_b, pc, psqt, out);
}
